// FP8Linear_11003706212931
// MI455X (gfx1250) — 8-bit matrix operand order repaired and checked by reading, compiled, not run on hardware
//
#include <hip/hip_runtime.h>
#include <hip/hip_bf16.h>
#include <hip/hip_fp16.h>


typedef __attribute__((ext_vector_type(16))) int   v16i;
typedef __attribute__((ext_vector_type(8)))  float v8f;
typedef __attribute__((ext_vector_type(4)))  int   vi4;

#define FP8_MAX_F 448.0f
#define SCALE_EPS 1e-6f

#if __has_builtin(__builtin_amdgcn_global_load_async_to_lds_b128)
#define USE_ASYNC_LDS 1
#else
#define USE_ASYNC_LDS 0
#endif

#if USE_ASYNC_LDS
typedef __attribute__((address_space(1))) void  gvoid_t;
typedef __attribute__((address_space(3))) void  lvoid_t;
typedef __attribute__((address_space(1))) vi4   gvi4_t;
typedef __attribute__((address_space(3))) vi4   lvi4_t;
#endif

__device__ __forceinline__ void copy16_g2l(const unsigned char* g, unsigned char* l) {
#if USE_ASYNC_LDS
  gvoid_t* gp = (gvoid_t*)(void*)g;
  lvoid_t* lp = (lvoid_t*)(void*)l;
  __builtin_amdgcn_global_load_async_to_lds_b128((gvi4_t*)gp, (lvi4_t*)lp, 0, 0);
#else
  *reinterpret_cast<int4*>(l) = *reinterpret_cast<const int4*>(g);
#endif
}

__device__ __forceinline__ void wait_async_prev() {
#if USE_ASYNC_LDS
#if __has_builtin(__builtin_amdgcn_s_wait_asynccnt)
  __builtin_amdgcn_s_wait_asynccnt(8);
#else
  asm volatile("s_wait_asynccnt 0x8" ::: "memory");
#endif
#endif
}

__device__ __forceinline__ void wait_async_all() {
#if USE_ASYNC_LDS
#if __has_builtin(__builtin_amdgcn_s_wait_asynccnt)
  __builtin_amdgcn_s_wait_asynccnt(0);
#else
  asm volatile("s_wait_asynccnt 0x0" ::: "memory");
#endif
#endif
}


__device__ __forceinline__ unsigned char f32_to_e4m3_manual(float f) {
  const unsigned char s = (__float_as_uint(f) >> 31) ? 0x80u : 0x00u;
  float a = fabsf(f);
  if (!(a < FP8_MAX_F)) a = FP8_MAX_F;
  if (a < 0.0009765625f) return s;
  int e; (void)frexpf(a, &e);
  int E = e + 6;
  if (E < 1) E = 1;
  int mi = (int)(ldexpf(a, 10 - E) + 0.5f);
  if (mi >= 16) { mi >>= 1; ++E; }
  if (E > 15) { E = 15; mi = 14; }
  return (unsigned char)(s | (mi >= 8 ? ((E << 3) | (mi - 8)) : mi));
}

__device__ __forceinline__ unsigned char f32_to_e4m3(float f) {
  f = fminf(fmaxf(f, -FP8_MAX_F), FP8_MAX_F);
#if __has_builtin(__builtin_amdgcn_cvt_pk_fp8_f32)
  return (unsigned char)(__builtin_amdgcn_cvt_pk_fp8_f32(f, f, 0, false) & 0xff);
#else
  return f32_to_e4m3_manual(f);
#endif
}


__global__ void __launch_bounds__(256)
quant_rows_kernel(const float* __restrict__ in,
                  unsigned char* __restrict__ out8,
                  float* __restrict__ scales,
                  int cols, int fp16_roundtrip) {
  const int row  = blockIdx.x;
  const int per  = cols >> 8;
  const int base = threadIdx.x * per;
  const float* rin = in + (size_t)row * cols;

  float4 vals[4];
  float amax = 0.0f;
#pragma unroll
  for (int g = 0; g < 4; ++g) {
    vals[g] = *reinterpret_cast<const float4*>(rin + base + g * 4);
    amax = fmaxf(amax, fmaxf(fmaxf(fabsf(vals[g].x), fabsf(vals[g].y)),
                             fmaxf(fabsf(vals[g].z), fabsf(vals[g].w))));
  }

  __shared__ float red[256];
  __shared__ float s_scale;
  red[threadIdx.x] = amax;
  __syncthreads();
#pragma unroll
  for (int s = 128; s > 0; s >>= 1) {
    if (threadIdx.x < s)
      red[threadIdx.x] = fmaxf(red[threadIdx.x], red[threadIdx.x + s]);
    __syncthreads();
  }
  if (threadIdx.x == 0) {
    float sc = fmaxf(red[0] / FP8_MAX_F, SCALE_EPS);
    if (fp16_roundtrip) sc = __half2float(__float2half(sc));
    s_scale = sc;
    scales[row] = sc;
  }
  __syncthreads();

  const float sc = s_scale;
  unsigned int packed[4];
#pragma unroll
  for (int g = 0; g < 4; ++g) {
    const unsigned int b0 = f32_to_e4m3(vals[g].x / sc);
    const unsigned int b1 = f32_to_e4m3(vals[g].y / sc);
    const unsigned int b2 = f32_to_e4m3(vals[g].z / sc);
    const unsigned int b3 = f32_to_e4m3(vals[g].w / sc);
    packed[g] = b0 | (b1 << 8) | (b2 << 16) | (b3 << 24);
  }
  *reinterpret_cast<uint4*>(out8 + (size_t)row * cols + base) =
      make_uint4(packed[0], packed[1], packed[2], packed[3]);
}


static constexpr int BM = 128, BN = 128, BK = 128;

__global__ void __launch_bounds__(256)
fp8_gemm_kernel(const unsigned char* __restrict__ a8,
                const unsigned char* __restrict__ b8,
                const float* __restrict__ ascale,
                const float* __restrict__ wscale,
                const float* __restrict__ bias,
                float* __restrict__ out,
                int K, int N) {
  __shared__ __align__(16) unsigned char smA[2][BM * BK];
  __shared__ __align__(16) unsigned char smB[2][BN * BK];

  const int tid   = threadIdx.x;
  const int lane  = tid & 31;
  const int wave  = tid >> 5;
  const int lo    = lane & 15;
  const int hi    = lane >> 4;
  const int waveM = wave & 3;
  const int waveN = wave >> 2;
  const int mblock = blockIdx.y * BM;
  const int nblock = blockIdx.x * BN;

  const int srow = tid >> 3;
  const int scol = (tid & 7) * 16;

  v8f acc[2][4];
#pragma unroll
  for (int mt = 0; mt < 2; ++mt)
#pragma unroll
    for (int nb = 0; nb < 4; ++nb)
      acc[mt][nb] = (v8f){0.f,0.f,0.f,0.f,0.f,0.f,0.f,0.f};

  auto stage = [&](int buf, int k0) {
#pragma unroll
    for (int c = 0; c < 4; ++c) {
      const int r = srow + c * 32;
      copy16_g2l(a8 + (size_t)(mblock + r) * K + k0 + scol, &smA[buf][r * BK + scol]);
      copy16_g2l(b8 + (size_t)(nblock + r) * K + k0 + scol, &smB[buf][r * BK + scol]);
    }
  };

  auto compute = [&](int buf) {
    union { v16i v; int i[16]; } Af[2];
#pragma unroll
    for (int mt = 0; mt < 2; ++mt) {
      const unsigned char* arow = &smA[buf][(waveM * 32 + mt * 16 + lo) * BK];
      const int ak = hi * 8;
#pragma unroll
      for (int j = 0; j < 8; ++j) {
        const int koff = ak * 2 + (j >> 1) * 32 + (j & 1) * 8;
        const int2 d = *reinterpret_cast<const int2*>(arow + koff);
        Af[mt].i[2 * j]     = d.x;
        Af[mt].i[2 * j + 1] = d.y;
      }
    }
    union { v16i v; int4 q[4]; } Bf[4];
#pragma unroll
    for (int nb = 0; nb < 4; ++nb) {
      const unsigned char* brow = &smB[buf][(waveN * 64 + nb * 16 + lo) * BK];
      const int bk = hi * 16;
#pragma unroll
      for (int j = 0; j < 4; ++j)
        Bf[nb].q[j] = *reinterpret_cast<const int4*>(brow + bk + j * 32);
    }
#pragma unroll
    for (int mt = 0; mt < 2; ++mt)
#pragma unroll
      for (int nb = 0; nb < 4; ++nb)
        acc[mt][nb] = __builtin_amdgcn_wmma_f32_16x16x128_fp8_fp8(
            Af[mt].v, Bf[nb].v, (short)0, acc[mt][nb],
             false,  false);
  };

  const int NK = K / BK;
  stage(0, 0);
  for (int ks = 0; ks < NK; ++ks) {
    const int cur = ks & 1;
    if (ks + 1 < NK) {
      stage(cur ^ 1, (ks + 1) * BK);
      wait_async_prev();
    } else {
      wait_async_all();
    }
    __syncthreads();
    compute(cur);
    __syncthreads();
  }

#pragma unroll
  for (int mt = 0; mt < 2; ++mt) {
#pragma unroll
    for (int nb = 0; nb < 4; ++nb) {
      const int n  = nblock + waveN * 64 + nb * 16 + lo;
      const float ws = wscale[n];
      const float bb = __bfloat162float(__float2bfloat16(bias[n]));
#pragma unroll
      for (int r = 0; r < 8; ++r) {
        const int m = mblock + waveM * 32 + mt * 16 + r + hi * 8;
        float o = acc[mt][nb][r] * ascale[m] * ws + bb;
        out[(size_t)m * N + n] = __bfloat162float(__float2bfloat16(o));
      }
    }
  }
}


extern "C" void kernel_launch(void* const* d_in, const int* in_sizes, int n_in,
                              void* d_out, int out_size, void* d_ws, size_t ws_size,
                              hipStream_t stream) {
  (void)n_in; (void)out_size; (void)ws_size;

  const float* x    = (const float*)d_in[0];
  const float* w    = (const float*)d_in[1];
  const float* bias = (const float*)d_in[2];

  const int d_outf = in_sizes[2];
  const int d_inf  = in_sizes[1] / d_outf;
  const int T      = in_sizes[0] / d_inf;

  float* out = (float*)d_out;

  unsigned char* base = (unsigned char*)d_ws;
  unsigned char* w8 = base;
  unsigned char* x8 = w8 + (size_t)d_outf * d_inf;
  size_t off = (size_t)d_outf * d_inf + (size_t)T * d_inf;
  off = (off + 255) & ~(size_t)255;
  float* wsc = (float*)(base + off);
  float* asc = wsc + d_outf;

  quant_rows_kernel<<<d_outf, 256, 0, stream>>>(w, w8, wsc, d_inf,  1);
  quant_rows_kernel<<<T,      256, 0, stream>>>(x, x8, asc, d_inf,  0);

  dim3 grid(d_outf / BN, T / BM);
  fp8_gemm_kernel<<<grid, 256, 0, stream>>>(x8, w8, asc, wsc, bias, out,
                                            d_inf, d_outf);
}
